// Attention_88502096101557
// MI455X (gfx1250) — compile-verified, matrix operand order checked by reading, not run on hardware
//
#include <hip/hip_runtime.h>

typedef __bf16 bf16;
typedef __attribute__((ext_vector_type(4)))  __bf16 v4bf;
typedef __attribute__((ext_vector_type(8)))  __bf16 v8bf;
typedef __attribute__((ext_vector_type(16))) __bf16 v16bf;
typedef __attribute__((ext_vector_type(8)))  float  v8f;

#define HID    4096
#define NHEAD  32
#define NKV    8
#define HDIM   128
#define BATCH  2
#define SEQ    2048
#define MROWS  (BATCH * SEQ)        // 4096 rows (B*L)
#define KVW    (2 * NKV * HDIM)     // 2048
#define KDW    (NKV * HDIM)         // 1024

// ---------------------------------------------------------------------------
// WMMA helpers (CDNA5 wave32, V_WMMA_F32_16X16X32_BF16)
// A-fragment layout (16x32 MxK, ISA 7.12.2): lane l<16 -> row l, K {0..7,16..23};
// lane l>=16 -> row l-16, K {8..15,24..31}. B-fragment = same layout applied to
// B^T (N x K row-major). C/D: lane = N, VGPR r = M (r, or r+8 for lanes 16..31).
// ---------------------------------------------------------------------------
__device__ __forceinline__ v16bf load_frag(const bf16* base, int ld) {
  const int lane = threadIdx.x & 31;
  const int row  = lane & 15;
  const int koff = (lane & 16) >> 1;           // 0 or 8
  const bf16* p = base + (size_t)row * ld + koff;
  v8bf lo = *reinterpret_cast<const v8bf*>(p);       // K +0..7   (b128)
  v8bf hi = *reinterpret_cast<const v8bf*>(p + 16);  // K +16..23 (b128)
  v16bf f;
#pragma unroll
  for (int i = 0; i < 8; ++i) { f[i] = lo[i]; f[i + 8] = hi[i]; }
  return f;
}

__device__ __forceinline__ v8f wmma_bf16(v16bf a, v16bf b, v8f c) {
  return __builtin_amdgcn_wmma_f32_16x16x32_bf16(false, a, false, b, (short)0, c,
                                                 false, false);
}

// Async global->LDS copy (GVS mode: SGPR 64-bit base + per-lane VGPR byte offset)
__device__ __forceinline__ void async_copy_b128(unsigned int lds_byte_addr,
                                                unsigned long long gbase,
                                                unsigned int gbyte_off) {
  asm volatile("global_load_async_to_lds_b128 %0, %1, %2"
               :: "v"(lds_byte_addr), "v"(gbyte_off), "s"(gbase)
               : "memory");
}

// ---------------------------------------------------------------------------
// fp32 -> bf16 elementwise convert (4 elems / thread)
// ---------------------------------------------------------------------------
__global__ void cvt_f32_bf16(const float* __restrict__ in, bf16* __restrict__ out,
                             int n4) {
  const int i = blockIdx.x * blockDim.x + threadIdx.x;
  if (i < n4) {
    const float4 v = reinterpret_cast<const float4*>(in)[i];
    v4bf o;
    o[0] = (bf16)v.x; o[1] = (bf16)v.y; o[2] = (bf16)v.z; o[3] = (bf16)v.w;
    reinterpret_cast<v4bf*>(out)[i] = o;
  }
}

// ---------------------------------------------------------------------------
// fp32 [K][N] -> bf16 [N][K] transpose+convert (LDS 32x32 tile, block 32x8)
// ---------------------------------------------------------------------------
__global__ void transpose_cvt(const float* __restrict__ W, bf16* __restrict__ Wt,
                              int K, int N) {
  __shared__ float tile[32][33];
  const int n0 = blockIdx.x * 32, k0 = blockIdx.y * 32;
  const int tx = threadIdx.x, ty = threadIdx.y;
#pragma unroll
  for (int i = 0; i < 32; i += 8)
    tile[ty + i][tx] = W[(size_t)(k0 + ty + i) * N + n0 + tx];
  __syncthreads();
#pragma unroll
  for (int i = 0; i < 32; i += 8)
    Wt[(size_t)(n0 + ty + i) * K + k0 + tx] = (bf16)tile[tx][ty + i];
}

// ---------------------------------------------------------------------------
// Tiled WMMA GEMM: C[M,N] = A[M,K] @ Bt[N,K]^T + bias.
// One wave per block; each wave computes a 32x64 strip:
// 2 A-frags x 4 B-frags -> 8 WMMAs per 12 b128 loads per k-step.
// MODE 0: bf16 out.  MODE 1: KV split epilogue.  MODE 2: fp32 out.
// ---------------------------------------------------------------------------
template <int MODE>
__global__ __launch_bounds__(32) void gemm_wmma(
    const bf16* __restrict__ A, const bf16* __restrict__ Bt,
    const float* __restrict__ bias, void* __restrict__ C0,
    bf16* __restrict__ C1, int N, int K) {
  const int n0 = blockIdx.x * 64;
  const int m0 = blockIdx.y * 32;
  v8f acc[2][4] = {};
  for (int k = 0; k < K; k += 32) {
    const v16bf a0 = load_frag(A + (size_t)m0 * K + k, K);
    const v16bf a1 = load_frag(A + (size_t)(m0 + 16) * K + k, K);
#pragma unroll
    for (int t = 0; t < 4; ++t) {
      const v16bf b = load_frag(Bt + (size_t)(n0 + 16 * t) * K + k, K);
      acc[0][t] = wmma_bf16(a0, b, acc[0][t]);
      acc[1][t] = wmma_bf16(a1, b, acc[1][t]);
    }
  }
  const int lane = threadIdx.x & 31;
  const int nl = lane & 15;
  const int rb = (lane & 16) >> 1;  // 0 or 8
#pragma unroll
  for (int u = 0; u < 2; ++u) {
#pragma unroll
    for (int t = 0; t < 4; ++t) {
      const int n = n0 + 16 * t + nl;
      const float bv = bias[n];
#pragma unroll
      for (int r = 0; r < 8; ++r) {
        const int m = m0 + 16 * u + rb + r;
        const float v = acc[u][t][r] + bv;
        if (MODE == 0) {
          ((bf16*)C0)[(size_t)m * N + n] = (bf16)v;
        } else if (MODE == 2) {
          ((float*)C0)[(size_t)m * N + n] = v;
        } else {  // KV split: n<1024 -> K[b,l,kvh,d]; n>=1024 -> V^T[b,kvh,d,l]
          if (n < KDW) {
            ((bf16*)C0)[(size_t)m * KDW + n] = (bf16)v;
          } else {
            const int c = n - KDW;              // kvh*128 + d
            const int b_ = m / SEQ, l = m % SEQ;
            C1[((size_t)(b_ * KDW + c)) * SEQ + l] = (bf16)v;
          }
        }
      }
    }
  }
}

// ---------------------------------------------------------------------------
// Flash attention. 8 waves per workgroup handle 128 queries of one head.
// 64-key rounds: K (64x128) and V^T (128x64) tiles staged into LDS with
// global_load_async_to_lds_b128 (double-buffered, s_wait_asynccnt drain),
// shared by all 8 waves. One cross-lane softmax reduction per 64 keys.
//   Q : [B*L, 4096] head-major; K : [B*L, 1024] kvh-major; Vt : [B*1024, L].
// Row paddings (136 / 72 elems) keep 16B alignment and make the 16-lane
// ds_load_b128 fragment reads bank-conflict-free.
// ---------------------------------------------------------------------------
#define KPAD 136
#define VPAD 72

__global__ __launch_bounds__(256) void flash_attn(
    const bf16* __restrict__ Q, const bf16* __restrict__ Kb,
    const bf16* __restrict__ Vt, bf16* __restrict__ O) {
  __shared__ __align__(16) bf16 Ks[2][64][KPAD];
  __shared__ __align__(16) bf16 Vs[2][128][VPAD];
  __shared__ __align__(16) bf16 Plds[8][16][VPAD];

  const int tid = threadIdx.x;
  const int wave = tid >> 5;
  const int lane = tid & 31;
  const int nl = lane & 15;
  const int rb = (lane & 16) >> 1;
  const int qb = blockIdx.x, h = blockIdx.y, b = blockIdx.z;
  const int kvh = h >> 2;                      // G = 4 query heads per KV head
  const size_t qrow0 = (size_t)b * SEQ + (size_t)qb * 128 + (size_t)wave * 16;

  const bf16* Kbase = Kb + (size_t)(b * SEQ) * KDW + kvh * HDIM;
  const bf16* Vbase = Vt + (size_t)(b * KDW + kvh * HDIM) * SEQ;
  const unsigned int ks_lds = (unsigned int)(uintptr_t)(&Ks[0][0][0]);
  const unsigned int vs_lds = (unsigned int)(uintptr_t)(&Vs[0][0][0]);

  // stage one 64-key K tile + V tile into buffer `buf` (8 async b128 / thread)
  auto issue_tiles = [&](int buf, int j) {
    const unsigned long long kg =
        (unsigned long long)(uintptr_t)(Kbase + (size_t)j * KDW);
    const unsigned long long vg = (unsigned long long)(uintptr_t)(Vbase + j);
#pragma unroll
    for (int q = 0; q < 4; ++q) {
      const int s = tid + q * 256;
      {  // K: 1024 slots of 16B -> Ks[buf][r][c*8..]
        const unsigned int r = s >> 4, c = s & 15;
        async_copy_b128(ks_lds + ((buf * 64 + r) * KPAD + c * 8) * 2, kg,
                        (r * KDW + c * 8) * 2);
      }
      {  // V: 1024 slots of 16B -> Vs[buf][d][c*8..]
        const unsigned int d = s >> 3, c = s & 7;
        async_copy_b128(vs_lds + ((buf * 128 + d) * VPAD + c * 8) * 2, vg,
                        (d * SEQ + c * 8) * 2);
      }
    }
  };

  issue_tiles(0, 0);

  v16bf qf[4];
  const bf16* Qbase = Q + qrow0 * (NHEAD * HDIM) + h * HDIM;
#pragma unroll
  for (int i = 0; i < 4; ++i) qf[i] = load_frag(Qbase + 32 * i, NHEAD * HDIM);

  v8f o[8] = {};
  float mi[8], li[8];
#pragma unroll
  for (int r = 0; r < 8; ++r) { mi[r] = -3.0e30f; li[r] = 0.0f; }
  const float scale = 0.08838834764831845f;    // 1/sqrt(128)

  const int NSTEPS = SEQ / 64;
  for (int jj = 0; jj < NSTEPS; ++jj) {
    const int buf = jj & 1;
    if (jj + 1 < NSTEPS) {
      issue_tiles(buf ^ 1, (jj + 1) * 64);
      asm volatile("s_wait_asynccnt 0x8" ::: "memory");  // older buffer done
    } else {
      asm volatile("s_wait_asynccnt 0x0" ::: "memory");
    }
    __syncthreads();

    // S = Q @ K^T for 64 keys (four 16x16 tiles), from LDS
    v8f s[4] = {};
#pragma unroll
    for (int t4 = 0; t4 < 4; ++t4) {
#pragma unroll
      for (int kk = 0; kk < 4; ++kk) {
        const v16bf kf = load_frag(&Ks[buf][t4 * 16][kk * 32], KPAD);
        s[t4] = wmma_bf16(qf[kk], kf, s[t4]);
      }
    }

    // online softmax: one cross-lane reduction per row per 64 keys
    float alpha[8];
#pragma unroll
    for (int r = 0; r < 8; ++r) {
      float sr[4];
#pragma unroll
      for (int i = 0; i < 4; ++i) sr[i] = s[i][r] * scale;
      float mrow = fmaxf(fmaxf(sr[0], sr[1]), fmaxf(sr[2], sr[3]));
#pragma unroll
      for (int msk = 1; msk < 16; msk <<= 1)
        mrow = fmaxf(mrow, __shfl_xor(mrow, msk, 32));
      const float mnew = fmaxf(mi[r], mrow);
      float p[4], rs = 0.0f;
#pragma unroll
      for (int i = 0; i < 4; ++i) { p[i] = __expf(sr[i] - mnew); rs += p[i]; }
#pragma unroll
      for (int msk = 1; msk < 16; msk <<= 1) rs += __shfl_xor(rs, msk, 32);
      alpha[r] = __expf(mi[r] - mnew);
      li[r] = li[r] * alpha[r] + rs;
      mi[r] = mnew;
#pragma unroll
      for (int i = 0; i < 4; ++i)
        Plds[wave][rb + r][i * 16 + nl] = (bf16)p[i];  // C-layout -> row-major
    }
#pragma unroll
    for (int t = 0; t < 8; ++t)
#pragma unroll
      for (int r = 0; r < 8; ++r) o[t][r] *= alpha[r];

    // same-wave DS ordering guarantees store->load; drain + compiler fence
    asm volatile("s_wait_dscnt 0x0" ::: "memory");
    const v16bf pf0 = load_frag(&Plds[wave][0][0], VPAD);
    const v16bf pf1 = load_frag(&Plds[wave][0][32], VPAD);
#pragma unroll
    for (int t = 0; t < 8; ++t) {
      const v16bf vf0 = load_frag(&Vs[buf][t * 16][0], VPAD);
      const v16bf vf1 = load_frag(&Vs[buf][t * 16][32], VPAD);
      o[t] = wmma_bf16(pf0, vf0, o[t]);
      o[t] = wmma_bf16(pf1, vf1, o[t]);
    }
    __syncthreads();
  }

#pragma unroll
  for (int r = 0; r < 8; ++r) {
    const float inv = 1.0f / li[r];
    const size_t m = qrow0 + rb + r;
#pragma unroll
    for (int t = 0; t < 8; ++t)
      O[m * (NHEAD * HDIM) + h * HDIM + t * 16 + nl] = (bf16)(o[t][r] * inv);
  }
}

// ---------------------------------------------------------------------------
extern "C" void kernel_launch(void* const* d_in, const int* in_sizes, int n_in,
                              void* d_out, int out_size, void* d_ws, size_t ws_size,
                              hipStream_t stream) {
  (void)in_sizes; (void)n_in; (void)out_size; (void)ws_size;
  const float* query = (const float*)d_in[0];
  const float* kv    = (const float*)d_in[1];
  const float* Wq    = (const float*)d_in[2];
  const float* bq    = (const float*)d_in[3];
  const float* Wkv   = (const float*)d_in[4];
  const float* bkv   = (const float*)d_in[5];
  const float* Wo    = (const float*)d_in[6];
  const float* bo    = (const float*)d_in[7];
  float* out = (float*)d_out;

  char* ws = (char*)d_ws;
  auto alloc = [&](size_t bytes) -> char* {
    char* p = ws;
    ws += (bytes + 255) & ~(size_t)255;
    return p;
  };
  bf16* qx   = (bf16*)alloc((size_t)MROWS * HID * 2);
  bf16* kvx  = (bf16*)alloc((size_t)MROWS * HID * 2);
  bf16* WqT  = (bf16*)alloc((size_t)HID * HID * 2);
  bf16* WkvT = (bf16*)alloc((size_t)KVW * HID * 2);
  bf16* WoT  = (bf16*)alloc((size_t)HID * HID * 2);
  bf16* Qp   = (bf16*)alloc((size_t)MROWS * (NHEAD * HDIM) * 2);
  bf16* Kbf  = (bf16*)alloc((size_t)MROWS * KDW * 2);
  bf16* Vt   = (bf16*)alloc((size_t)BATCH * KDW * SEQ * 2);
  bf16* Ao   = (bf16*)alloc((size_t)MROWS * (NHEAD * HDIM) * 2);

  // activations fp32 -> bf16
  {
    const int n4 = MROWS * HID / 4;
    cvt_f32_bf16<<<(n4 + 255) / 256, 256, 0, stream>>>(query, qx, n4);
    cvt_f32_bf16<<<(n4 + 255) / 256, 256, 0, stream>>>(kv, kvx, n4);
  }
  // weights fp32 [K][N] -> bf16 [N][K]
  transpose_cvt<<<dim3(HID / 32, HID / 32), dim3(32, 8), 0, stream>>>(Wq, WqT, HID, HID);
  transpose_cvt<<<dim3(KVW / 32, HID / 32), dim3(32, 8), 0, stream>>>(Wkv, WkvT, HID, KVW);
  transpose_cvt<<<dim3(HID / 32, HID / 32), dim3(32, 8), 0, stream>>>(Wo, WoT, HID, HID);

  // Q projection -> bf16 [M, 4096] (head-major columns)
  gemm_wmma<0><<<dim3((NHEAD * HDIM) / 64, MROWS / 32), 32, 0, stream>>>(
      qx, WqT, bq, (void*)Qp, nullptr, NHEAD * HDIM, HID);
  // KV projection -> K [M, 1024] + V^T [B*1024, L]
  gemm_wmma<1><<<dim3(KVW / 64, MROWS / 32), 32, 0, stream>>>(
      kvx, WkvT, bkv, (void*)Kbf, Vt, KVW, HID);
  // flash attention -> bf16 [M, 4096]
  flash_attn<<<dim3(SEQ / 128, NHEAD, BATCH), 256, 0, stream>>>(Qp, Kbf, Vt, Ao);
  // O projection -> fp32 output
  gemm_wmma<2><<<dim3(HID / 64, MROWS / 32), 32, 0, stream>>>(
      Ao, WoT, bo, (void*)out, nullptr, HID, HID);
}
